// GNNEncoder_44573170598349
// MI455X (gfx1250) — matrix operand order repaired and checked by reading, not compiled, not run on hardware
//
#include <hip/hip_runtime.h>
#include <hip/hip_bf16.h>

#define NN 50000
#define NE 800000
#define NG 128
#define HIDF 64
#define LATF 128
#define PROJF 32
#define NTILES (NE / 16)   // 50000 exactly, no tail

typedef __attribute__((ext_vector_type(16))) __bf16       v16bf;
typedef __attribute__((ext_vector_type(8)))  float        v8f;
typedef __attribute__((ext_vector_type(8)))  unsigned int v8u;

// ---------- helpers ----------
__device__ __forceinline__ unsigned short f2bf(float f) {
    unsigned int u = __builtin_bit_cast(unsigned int, f);
    u += 0x7fffu + ((u >> 16) & 1u);            // round-to-nearest-even
    return (unsigned short)(u >> 16);
}
__device__ __forceinline__ float bf2f(unsigned int h) {
    return __builtin_bit_cast(float, (h & 0xffffu) << 16);
}
__device__ __forceinline__ unsigned int pk2(float a, float b) {
    return (unsigned int)f2bf(a) | ((unsigned int)f2bf(b) << 16);
}
// monotonic float<->uint map so atomicMax(u32) == float max
__device__ __forceinline__ unsigned int encf(float f) {
    unsigned int u = __builtin_bit_cast(unsigned int, f);
    return (u & 0x80000000u) ? ~u : (u | 0x80000000u);
}
__device__ __forceinline__ float decf(unsigned int e) {
    unsigned int u = (e & 0x80000000u) ? (e & 0x7fffffffu) : ~e;
    return __builtin_bit_cast(float, u);
}

// ---------- per-edge MLP + scatter-max (the hot kernel) ----------
// One wave handles a tile of 16 edges. GEMM1: [16 x KT*32] x [KT*32 x 64]
// (bf16 WMMA, f32 acc), bias+relu, then GEMM2: [16 x 64] x [64 x 64],
// bias, then encoded atomic-max scatter to agg[dst][64].
template <int KT, bool FIRST>
__global__ __launch_bounds__(256) void edge_conv_kernel(
    const float* __restrict__ xin,            // FIRST: raw x [NN,4] fp32
    const unsigned short* __restrict__ feat,  // !FIRST: node feats [NN,64] bf16
    const int* __restrict__ eidx,             // [2, NE] (src row, dst row)
    const float* __restrict__ W1, const float* __restrict__ B1,
    const float* __restrict__ W2, const float* __restrict__ B2,
    unsigned int* __restrict__ agg)           // [NN,64] encoded max
{
    __shared__ v8u w1pk[KT * 4][32];          // B-fragments of W1: [kt*4+nt][lane]
    __shared__ v8u w2pk[8][32];               // B-fragments of W2
    __shared__ float b1s[HIDF], b2s[HIDF];
    __shared__ unsigned short hid[8][16 * HIDF]; // per-wave hidden tile (bf16)

    const int tid = threadIdx.x;
    const int INK = FIRST ? 8 : 128;

    // pack W1 into WMMA B-fragment layout (zero-pad K beyond INK)
    unsigned int* w1u = (unsigned int*)w1pk;
    for (int idx = tid; idx < KT * 4 * 256; idx += 256) {
        int v = idx & 7, lane = (idx >> 3) & 31, f = idx >> 8;
        int kt = f >> 2, nt = f & 3;
        int n = nt * 16 + (lane & 15);
        int k = kt * 32 + 2 * v + (v >= 4 ? 8 : 0) + 8 * (lane >> 4);   // B layout: K=2v+(v>=4?8:0)+8*half
        float f0 = (k     < INK) ? W1[k * HIDF + n]       : 0.f;
        float f1 = (k + 1 < INK) ? W1[(k + 1) * HIDF + n] : 0.f;
        w1u[idx] = pk2(f0, f1);
    }
    unsigned int* w2u = (unsigned int*)w2pk;
    for (int idx = tid; idx < 8 * 256; idx += 256) {
        int v = idx & 7, lane = (idx >> 3) & 31, f = idx >> 8;
        int kt = f >> 2, nt = f & 3;
        int n = nt * 16 + (lane & 15);
        int k = kt * 32 + 2 * v + (v >= 4 ? 8 : 0) + 8 * (lane >> 4);
        w2u[idx] = pk2(W2[k * HIDF + n], W2[(k + 1) * HIDF + n]);
    }
    if (tid < HIDF) { b1s[tid] = B1[tid]; b2s[tid] = B2[tid]; }
    __syncthreads();

    const int wv = tid >> 5, lane = tid & 31;
    const int m = lane & 15, half = lane >> 4, nb = lane & 15;
    unsigned short* hrow = hid[wv];

    for (int tile = blockIdx.x * 8 + wv; tile < NTILES; tile += gridDim.x * 8) {
        const int ebase = tile * 16;
        const int e  = ebase + m;
        const int sN = eidx[e];
        const int dN = eidx[NE + e];

        v8f acc[4];
        for (int nt = 0; nt < 4; ++nt)
            for (int r = 0; r < 8; ++r) acc[nt][r] = 0.f;

        // ---- GEMM1: msg-input [x_i, x_j - x_i] ----
        for (int kt = 0; kt < KT; ++kt) {
            v8u ap;
            #pragma unroll
            for (int v = 0; v < 8; ++v) {
                // A layout: K pair = 2v + (v>=4 ? 8:0) + 8*half
                int k0 = kt * 32 + 2 * v + (v >= 4 ? 8 : 0) + 8 * half;
                unsigned int val;
                if (FIRST) {
                    float a0, a1;
                    if (k0 < 4)      { a0 = xin[dN * 4 + k0];  a1 = xin[dN * 4 + k0 + 1]; }
                    else if (k0 < 8) { int q = k0 - 4;
                                       a0 = xin[sN * 4 + q]     - xin[dN * 4 + q];
                                       a1 = xin[sN * 4 + q + 1] - xin[dN * 4 + q + 1]; }
                    else             { a0 = 0.f; a1 = 0.f; }
                    val = pk2(a0, a1);
                } else {
                    if (k0 < HIDF) {         // x_i half: direct packed bf16 pair
                        val = *(const unsigned int*)(feat + dN * HIDF + k0);
                    } else {                 // (x_j - x_i) half
                        int q = k0 - HIDF;
                        unsigned int ui = *(const unsigned int*)(feat + dN * HIDF + q);
                        unsigned int uj = *(const unsigned int*)(feat + sN * HIDF + q);
                        val = pk2(bf2f(uj) - bf2f(ui), bf2f(uj >> 16) - bf2f(ui >> 16));
                    }
                }
                ap[v] = val;
            }
            v16bf a = __builtin_bit_cast(v16bf, ap);
            #pragma unroll
            for (int nt = 0; nt < 4; ++nt) {
                v16bf b = __builtin_bit_cast(v16bf, w1pk[kt * 4 + nt][lane]);
                acc[nt] = __builtin_amdgcn_wmma_f32_16x16x32_bf16(
                    false, a, false, b, (short)0, acc[nt], false, false);
            }
        }

        // bias + relu -> hidden tile in LDS (C layout -> row-major bf16)
        #pragma unroll
        for (int nt = 0; nt < 4; ++nt) {
            int n = nt * 16 + nb;
            float bb = b1s[n];
            #pragma unroll
            for (int r = 0; r < 8; ++r) {
                float vv = fmaxf(acc[nt][r] + bb, 0.f);
                hrow[(r + 8 * half) * HIDF + n] = f2bf(vv);   // m = r + 8*half
            }
        }
        __builtin_amdgcn_wave_barrier();   // DS in-order per wave + sched barrier

        // ---- GEMM2: hidden [16x64] x W2 [64x64] ----
        v8f acc2[4];
        for (int nt = 0; nt < 4; ++nt)
            for (int r = 0; r < 8; ++r) acc2[nt][r] = 0.f;
        #pragma unroll
        for (int kt = 0; kt < 2; ++kt) {
            v8u ap;
            #pragma unroll
            for (int v = 0; v < 8; ++v) {
                int k0 = kt * 32 + 2 * v + (v >= 4 ? 8 : 0) + 8 * half;
                ap[v] = *(const unsigned int*)&hrow[m * HIDF + k0];
            }
            v16bf a = __builtin_bit_cast(v16bf, ap);
            #pragma unroll
            for (int nt = 0; nt < 4; ++nt) {
                v16bf b = __builtin_bit_cast(v16bf, w2pk[kt * 4 + nt][lane]);
                acc2[nt] = __builtin_amdgcn_wmma_f32_16x16x32_bf16(
                    false, a, false, b, (short)0, acc2[nt], false, false);
            }
        }
        __builtin_amdgcn_wave_barrier();

        // ---- scatter: encoded atomic max into agg[dst][n] ----
        int dstm[8];
        #pragma unroll
        for (int r = 0; r < 8; ++r) dstm[r] = eidx[NE + ebase + r + 8 * half];
        #pragma unroll
        for (int nt = 0; nt < 4; ++nt) {
            int n = nt * 16 + nb;
            float bb = b2s[n];
            #pragma unroll
            for (int r = 0; r < 8; ++r)
                atomicMax(&agg[dstm[r] * HIDF + n], encf(acc2[nt][r] + bb));
        }
    }
}

// decode encoded max -> relu -> bf16 node features
__global__ __launch_bounds__(256) void decode_relu_kernel(
    const unsigned int* __restrict__ agg, unsigned short* __restrict__ feat, int n)
{
    int i = blockIdx.x * blockDim.x + threadIdx.x;
    if (i >= n) return;
    feat[i] = f2bf(fmaxf(decf(agg[i]), 0.f));
}

// decode conv3 output (no relu) and accumulate graph mean-pool sums
__global__ __launch_bounds__(256) void decode_pool_kernel(
    const unsigned int* __restrict__ agg, const int* __restrict__ batch,
    float* __restrict__ sums, float* __restrict__ cnt)
{
    int i = blockIdx.x * blockDim.x + threadIdx.x;
    if (i >= NN * HIDF) return;
    int node = i >> 6;
    int g = batch[node];
    atomicAdd(&sums[g * HIDF + (i & 63)], decf(agg[i]));
    if ((i & 63) == 0) atomicAdd(&cnt[g], 1.0f);
}

// mean pool + project MLP + head MLP (tiny: 128 rows, scalar fp32)
__global__ __launch_bounds__(128) void head_kernel(
    const float* __restrict__ sums, const float* __restrict__ cnt,
    const float* __restrict__ pw1, const float* __restrict__ pb1,
    const float* __restrict__ pw2, const float* __restrict__ pb2,
    const float* __restrict__ hw1, const float* __restrict__ hb1,
    const float* __restrict__ hw2, const float* __restrict__ hb2,
    float* __restrict__ out)
{
    __shared__ float pooled[HIDF], h1[LATF], zz[LATF], h2[LATF];
    int g = blockIdx.x, t = threadIdx.x;
    if (t < HIDF) pooled[t] = sums[g * HIDF + t] / fmaxf(cnt[g], 1.0f);
    __syncthreads();
    float a = pb1[t];
    for (int k = 0; k < HIDF; ++k) a += pooled[k] * pw1[k * LATF + t];
    h1[t] = fmaxf(a, 0.f);
    __syncthreads();
    float b = pb2[t];
    for (int k = 0; k < LATF; ++k) b += h1[k] * pw2[k * LATF + t];
    zz[t] = b;
    __syncthreads();
    float c = hb1[t];
    for (int k = 0; k < LATF; ++k) c += zz[k] * hw1[k * LATF + t];
    h2[t] = fmaxf(c, 0.f);
    __syncthreads();
    if (t < PROJF) {
        float d = hb2[t];
        for (int k = 0; k < LATF; ++k) d += h2[k] * hw2[k * PROJF + t];
        out[g * PROJF + t] = d;
    }
}

extern "C" void kernel_launch(void* const* d_in, const int* in_sizes, int n_in,
                              void* d_out, int out_size, void* d_ws, size_t ws_size,
                              hipStream_t stream)
{
    (void)out_size; (void)ws_size;
    // detect input ordering: insertion order (x first) vs sorted tree-flatten
    int ix, ie, ib, ip;
    if (in_sizes[0] == NN * 4) { ix = 0; ie = 1; ib = 2; ip = 3; }           // x, edge_index, batch, params
    else                       { ib = 0; ie = 1; ip = 2; ix = n_in - 1; }    // batch, edge_index, params..., x
    const float* x    = (const float*)d_in[ix];
    const int*   eidx = (const int*)d_in[ie];
    const int*   batch = (const int*)d_in[ib];
    const float* P[20];
    for (int i = 0; i < 20; ++i) P[i] = (const float*)d_in[ip + i];
    // P[0..3]=conv1, P[4..7]=conv2, P[8..11]=conv3; then project/head in either order
    const float *pw1, *pb1, *pw2, *pb2, *hw1, *hb1, *hw2, *hb2;
    if (in_sizes[ip + 12] == HIDF * LATF) {   // 8192 -> project.w1 first
        pw1 = P[12]; pb1 = P[13]; pw2 = P[14]; pb2 = P[15];
        hw1 = P[16]; hb1 = P[17]; hw2 = P[18]; hb2 = P[19];
    } else {                                  // 16384 -> head.w1 first (sorted keys)
        hw1 = P[12]; hb1 = P[13]; hw2 = P[14]; hb2 = P[15];
        pw1 = P[16]; pb1 = P[17]; pw2 = P[18]; pb2 = P[19];
    }

    char* ws = (char*)d_ws;
    unsigned int*   agg  = (unsigned int*)ws;                               // 12.8 MB
    unsigned short* feat = (unsigned short*)(ws + (size_t)NN * HIDF * 4);   // 6.4 MB
    float*          sums = (float*)(ws + (size_t)NN * HIDF * 6);            // 32 KB
    float*          cnt  = sums + NG * HIDF;                                // 512 B

    const int EB = 1250;                      // 10000 waves -> 5 tiles each
    const int DB = (NN * HIDF + 255) / 256;

    hipMemsetAsync(agg, 0, (size_t)NN * HIDF * 4, stream);
    hipMemsetAsync(sums, 0, (NG * HIDF + NG) * sizeof(float), stream);

    edge_conv_kernel<1, true><<<EB, 256, 0, stream>>>(
        x, nullptr, eidx, P[0], P[1], P[2], P[3], agg);
    decode_relu_kernel<<<DB, 256, 0, stream>>>(agg, feat, NN * HIDF);

    hipMemsetAsync(agg, 0, (size_t)NN * HIDF * 4, stream);
    edge_conv_kernel<4, false><<<EB, 256, 0, stream>>>(
        nullptr, feat, eidx, P[4], P[5], P[6], P[7], agg);
    decode_relu_kernel<<<DB, 256, 0, stream>>>(agg, feat, NN * HIDF);

    hipMemsetAsync(agg, 0, (size_t)NN * HIDF * 4, stream);
    edge_conv_kernel<4, false><<<EB, 256, 0, stream>>>(
        nullptr, feat, eidx, P[8], P[9], P[10], P[11], agg);
    decode_pool_kernel<<<DB, 256, 0, stream>>>(agg, batch, sums, cnt);

    head_kernel<<<NG, LATF, 0, stream>>>(
        sums, cnt, pw1, pb1, pw2, pb2, hw1, hb1, hw2, hb2, (float*)d_out);
}
